// DirModel_58523224375705
// MI455X (gfx1250) — matrix operand order repaired and checked by reading, not compiled, not run on hardware
//
#include <hip/hip_runtime.h>
#include <hip/hip_bf16.h>
#include <math.h>

// Problem constants (match reference)
#define B_  2
#define N_  1024
#define F__ 2048
#define CI_ 3
#define C_  80
#define CO_ 128
#define NB_ 15
#define Q_  20

// Split-K factor for the streaming (Dirac / L@u) GEMMs: raises workgroup count
// 4x so enough waves are resident to cover HBM latency at 23.3 TB/s.
#define KSPLIT 4

// Use gfx1250 async global->LDS copies (ASYNCcnt path) for B-panel staging.
// Fallback (0) is the proven plain-LDS-store path.
#define USE_ASYNC_STAGE 1

typedef __attribute__((ext_vector_type(16))) __bf16 v16bf;
typedef __attribute__((ext_vector_type(8)))  float  v8f;

static __device__ __forceinline__ unsigned short f32_bf16(float f) {
  unsigned int u = __float_as_uint(f);
  u += 0x7fffu + ((u >> 16) & 1u);      // round-to-nearest-even
  return (unsigned short)(u >> 16);
}
static __device__ __forceinline__ float eluf(float x) {
  return x > 0.f ? x : expm1f(x);
}

// ---------------- utility kernels ----------------
__global__ void k_zero_f32(float* p, long n) {
  long i = (long)blockIdx.x * blockDim.x + threadIdx.x;
  long s = (long)gridDim.x * blockDim.x;
  for (; i < n; i += s) p[i] = 0.f;
}

__global__ void k_cvt_bf16(const float* __restrict__ s, unsigned short* __restrict__ d, long n) {
  long i = (long)blockIdx.x * blockDim.x + threadIdx.x;
  long st = (long)gridDim.x * blockDim.x;
  for (; i < n; i += st) d[i] = f32_bf16(s[i]);
}

// dst[b][q][k] = q<nvalid ? bf16(src[b][k][q]) : 0   (B transposed + padded, bf16)
__global__ void k_transpose_bf16(const float* __restrict__ src, unsigned short* __restrict__ dst,
                                 int K, int nvalid, int npad, long sstride, long dstride) {
  int b = blockIdx.z;
  int k = blockIdx.x * blockDim.x + threadIdx.x;
  if (k >= K) return;
  const float* s = src + (long)b * sstride + (long)k * nvalid;
  unsigned short* d = dst + (long)b * dstride + k;
  for (int q = 0; q < npad; ++q)
    d[(long)q * K] = (q < nvalid) ? f32_bf16(s[q]) : (unsigned short)0;
}

// sum KS split-K partials: o[b][i] = sum_s p[(b*KS+s)][i]
__global__ void k_reduce_k(const float* __restrict__ p, float* __restrict__ o, long MN, int KS) {
  long idx = (long)blockIdx.x * blockDim.x + threadIdx.x;
  long tot = (long)B_ * MN;
  long stride = (long)gridDim.x * blockDim.x;
  for (; idx < tot; idx += stride) {
    long b = idx / MN, i = idx - b * MN;
    const float* q = p + (b * KS) * MN + i;
    float s = 0.f;
    for (int k = 0; k < KS; ++k) s += q[(long)k * MN];
    o[idx] = s;
  }
}

// ---------------- WMMA skinny GEMM with split-K ----------------
// Cpart[b*gridDim.y + seg][M, n_valid] = A_bf16[M, kBeg:kEnd] @ B (BT_bf16[NT*16, K])
template<int NT>
__global__ __launch_bounds__(256) void k_wmma_gemm(
    const unsigned short* __restrict__ A, const unsigned short* __restrict__ BT,
    float* __restrict__ Cpart, int K, int Kseg, int n_valid,
    long strideA, long strideBT, long strideCp)
{
  constexpr int CH = 256;
  __shared__ __align__(16) unsigned short lb[NT * 16][CH + 8]; // +8 pad: bank-conflict dodge
  int b = blockIdx.z;
  const unsigned short* Ab = A + (long)b * strideA;
  const unsigned short* Bb = BT + (long)b * strideBT;
  float* Cb = Cpart + (long)(b * gridDim.y + blockIdx.y) * strideCp;
  int kBeg = blockIdx.y * Kseg, kEnd = kBeg + Kseg;

  int t = threadIdx.x, lane = t & 31, hi = lane >> 4;
  int rowBase = blockIdx.x * 128 + (t >> 5) * 16;
  long aoff = (long)(rowBase + (lane & 15)) * K;

  v8f acc[NT];
#pragma unroll
  for (int i = 0; i < NT; ++i) acc[i] = (v8f){0.f,0.f,0.f,0.f,0.f,0.f,0.f,0.f};

  for (int kc = kBeg; kc < kEnd; kc += CH) {
    // cooperative stage of BT chunk (rows contiguous in global)
#pragma unroll
    for (int i = 0; i < NT * 4; ++i) {
      int e = (t + i * 256) * 4;
      int r = e / CH, c = e % CH;
#if USE_ASYNC_STAGE
      // gfx1250 async copy global -> LDS (b64 = 4 bf16), tracked by ASYNCcnt.
      // lb is the only __shared__ object -> group-segment offset 0.
      unsigned int ldsoff = (unsigned int)(r * (CH + 8) + c) * 2u;
      const unsigned short* gp = &Bb[(long)r * K + kc + c];
      asm volatile("global_load_async_to_lds_b64 %0, %1, off"
                   :: "v"(ldsoff), "v"(gp) : "memory");
#else
      *(uint2*)&lb[r][c] = *(const uint2*)&Bb[(long)r * K + kc + c];
#endif
    }
#if USE_ASYNC_STAGE
    asm volatile("s_wait_asynccnt 0x0" ::: "memory");
#endif
    __syncthreads();
    if (kc + CH < kEnd)
      __builtin_prefetch(Ab + aoff + kc + CH + hi * 8, 0, 0);  // global_prefetch_b8
#pragma unroll
    for (int kk = 0; kk < CH; kk += 32) {
      // A 16x32 bf16 fragment: lane<16 -> K kk..kk+7 & kk+16..kk+23 of row lane; lane>=16 -> +8
      union { v16bf v; uint4 q[2]; } af;
      int kb = kc + kk + hi * 8;
      af.q[0] = *(const uint4*)(Ab + aoff + kb);
      af.q[1] = *(const uint4*)(Ab + aoff + kb + 16);
#pragma unroll
      for (int nt = 0; nt < NT; ++nt) {
        // B 32x16 bf16 fragment: column lane&15, K kk..kk+7 & kk+16..kk+23 (lane>=16 -> +8)
        union { v16bf v; uint4 q[2]; } bfr;
        const unsigned short* bp = &lb[nt * 16 + (lane & 15)][kk + hi * 8];
        bfr.q[0] = *(const uint4*)bp;
        bfr.q[1] = *(const uint4*)(bp + 16);
        acc[nt] = __builtin_amdgcn_wmma_f32_16x16x32_bf16(
            false, af.v, false, bfr.v, (short)0, acc[nt], false, false);
      }
    }
    __syncthreads();
  }
#pragma unroll
  for (int nt = 0; nt < NT; ++nt) {
    int n = nt * 16 + (lane & 15);
    if (n < n_valid) {
#pragma unroll
      for (int r = 0; r < 8; ++r) {
        int row = rowBase + r + hi * 8;
        Cb[(long)row * n_valid + n] = acc[nt][r];
      }
    }
  }
}

// ---------------- per-channel mean/rstd over rows; channels 0..ncha-1 from a, rest from bsrc ----------------
__global__ void k_stats(const float* __restrict__ a, const float* __restrict__ bsrc,
                        int rows, int ncha, float* __restrict__ mu, float* __restrict__ rstd,
                        int off) {
  int c = blockIdx.x;
  const float* s = (c < ncha) ? (a + c) : (bsrc + (c - ncha));
  __shared__ float s1[256], s2[256];
  float x1 = 0.f, x2 = 0.f;
  for (int r = threadIdx.x; r < rows; r += 256) {
    float v = s[(long)r * 80];
    x1 += v; x2 += v * v;
  }
  s1[threadIdx.x] = x1; s2[threadIdx.x] = x2;
  __syncthreads();
  for (int o = 128; o > 0; o >>= 1) {
    if (threadIdx.x < o) { s1[threadIdx.x] += s1[threadIdx.x + o]; s2[threadIdx.x] += s2[threadIdx.x + o]; }
    __syncthreads();
  }
  if (threadIdx.x == 0) {
    float m = s1[0] / rows;
    float var = s2[0] / rows - m * m;
    mu[off + c] = m;
    rstd[off + c] = rsqrtf(var + 1e-5f);
  }
}

// fold BN into FC: WT[c][k] = bf16(W[k][c]*g[k]*rstd[k]);  be[c] = c0[c] + sum_k (b[k]-mu[k]*g[k]*rstd[k])*W[k][c]
__global__ void k_fold(const float* __restrict__ W, const float* __restrict__ g,
                       const float* __restrict__ bb, const float* __restrict__ mu,
                       const float* __restrict__ rstd, const float* __restrict__ c0,
                       unsigned short* __restrict__ WT, float* __restrict__ be) {
  int c = threadIdx.x;
  if (c >= 80) return;
  float acc = c0[c];
  for (int k = 0; k < 160; ++k) {
    float s = g[k] * rstd[k];
    float w = W[k * 80 + c];
    WT[c * 160 + k] = f32_bf16(w * s);
    acc += (bb[k] - mu[k] * s) * w;
  }
  be[c] = acc;
}

// ---------------- WMMA FC: out = elu([xa|xb] @ Weff + be) + addb, M x 160 @ 160 x 80 ----------------
__global__ __launch_bounds__(256) void k_wmma_fc(
    const float* __restrict__ xa, const float* __restrict__ xb,
    const unsigned short* __restrict__ WT, const float* __restrict__ be,
    const float* __restrict__ addb, float* __restrict__ out)
{
  __shared__ __align__(16) unsigned short lx[128][168];  // 128 rows x 160 ch (+8 pad)
  __shared__ __align__(16) unsigned short lw[80][168];   // 80 cols x 160 K (+8 pad)
  int t = threadIdx.x, lane = t & 31, wave = t >> 5, hi = lane >> 4;
  int rb = blockIdx.x * 128;

  for (int i = t; i < 3200; i += 256) {          // 80*160 bf16, vec4
    int e = i * 4, r = e / 160, c = e % 160;
    *(uint2*)&lw[r][c] = *(const uint2*)&WT[e];
  }
  for (int h = 0; h < 2; ++h) {
    const float* src = h ? xb : xa;
    for (int i = t; i < 10240; i += 256) {       // 128*80 fp32 -> bf16
      int r = i / 80, c = i % 80;
      lx[r][h * 80 + c] = f32_bf16(src[(long)(rb + r) * 80 + c]);
    }
  }
  __syncthreads();

  v8f acc[5];
#pragma unroll
  for (int i = 0; i < 5; ++i) acc[i] = (v8f){0.f,0.f,0.f,0.f,0.f,0.f,0.f,0.f};

#pragma unroll
  for (int kk = 0; kk < 160; kk += 32) {
    union { v16bf v; uint4 q[2]; } af;
    const unsigned short* ap = &lx[wave * 16 + (lane & 15)][kk + hi * 8];
    af.q[0] = *(const uint4*)ap;
    af.q[1] = *(const uint4*)(ap + 16);
#pragma unroll
    for (int nt = 0; nt < 5; ++nt) {
      union { v16bf v; uint4 q[2]; } bfr;
      const unsigned short* bp = &lw[nt * 16 + (lane & 15)][kk + hi * 8];
      bfr.q[0] = *(const uint4*)bp;
      bfr.q[1] = *(const uint4*)(bp + 16);
      acc[nt] = __builtin_amdgcn_wmma_f32_16x16x32_bf16(
          false, af.v, false, bfr.v, (short)0, acc[nt], false, false);
    }
  }
#pragma unroll
  for (int nt = 0; nt < 5; ++nt) {
    int n = nt * 16 + (lane & 15);
#pragma unroll
    for (int r = 0; r < 8; ++r) {
      int row = rb + wave * 16 + r + hi * 8;
      float v = eluf(acc[nt][r] + be[n]) + addb[(long)row * 80 + n];
      out[(long)row * 80 + n] = v;
    }
  }
}

// ---------------- conv1 / merge / conv2 / pool (small, fp32) ----------------
__global__ void k_Lx(const float* __restrict__ L, const float* __restrict__ x, float* __restrict__ Lx) {
  int idx = blockIdx.x * 256 + threadIdx.x;
  if (idx >= B_ * N_ * CI_) return;
  int ci = idx % CI_, n = (idx / CI_) % N_, b = idx / (CI_ * N_);
  const float* Lr = L + ((long)b * N_ + n) * N_;
  const float* xc = x + (long)b * N_ * CI_ + ci;
  float s = 0.f;
  for (int m = 0; m < N_; ++m) s += Lr[m] * xc[(long)m * CI_];
  Lx[idx] = s;
}

__global__ void k_conv1(const float* __restrict__ x, const float* __restrict__ Lx,
                        const float* __restrict__ W0, const float* __restrict__ W1,
                        const float* __restrict__ bias, float* __restrict__ v) {
  int idx = blockIdx.x * 256 + threadIdx.x;
  if (idx >= B_ * N_ * C_) return;
  int c = idx % C_; long bn = idx / C_;
  float s = bias[c];
  for (int ci = 0; ci < CI_; ++ci)
    s += x[bn * CI_ + ci] * W0[ci * C_ + c] + Lx[bn * CI_ + ci] * W1[ci * C_ + c];
  v[idx] = s;
}

__global__ void k_merge(const float* __restrict__ v2, const float* __restrict__ v1, const float* __restrict__ v0,
                        const float* __restrict__ mu, const float* __restrict__ rstd,
                        const float* __restrict__ g1, const float* __restrict__ b1,
                        const float* __restrict__ g2, const float* __restrict__ b2,
                        const float* __restrict__ g3, const float* __restrict__ b3,
                        float* __restrict__ u) {
  int idx = blockIdx.x * 256 + threadIdx.x;
  if (idx >= B_ * N_ * C_) return;
  int c = idx % C_;
  float s = (v2[idx] - mu[c])       * rstd[c]       * g1[c] + b1[c]
          + (v1[idx] - mu[80 + c])  * rstd[80 + c]  * g2[c] + b2[c]
          + (v0[idx] - mu[160 + c]) * rstd[160 + c] * g3[c] + b3[c];
  u[idx] = eluf(s);
}

__global__ void k_conv2(const float* __restrict__ u, const float* __restrict__ Lu,
                        const float* __restrict__ W0, const float* __restrict__ W1,
                        const float* __restrict__ bias, float* __restrict__ vo) {
  int idx = blockIdx.x * 256 + threadIdx.x;
  if (idx >= B_ * N_ * CO_) return;
  int co = idx % CO_; long bn = idx / CO_;
  float s = bias[co];
  for (int c = 0; c < C_; ++c)
    s += u[bn * C_ + c] * W0[c * CO_ + co] + Lu[bn * C_ + c] * W1[c * CO_ + co];
  vo[idx] = s;
}

__global__ void k_pool(const float* __restrict__ vo, const float* __restrict__ mask, float* __restrict__ out) {
  int co = blockIdx.x, b = blockIdx.y;
  __shared__ float s1[256], s2[256];
  float a = 0.f, m = 0.f;
  for (int n = threadIdx.x; n < N_; n += 256) {
    float mk = mask[b * N_ + n];
    a += vo[((long)b * N_ + n) * CO_ + co] * mk;
    m += mk;
  }
  s1[threadIdx.x] = a; s2[threadIdx.x] = m;
  __syncthreads();
  for (int o = 128; o > 0; o >>= 1) {
    if (threadIdx.x < o) { s1[threadIdx.x] += s1[threadIdx.x + o]; s2[threadIdx.x] += s2[threadIdx.x + o]; }
    __syncthreads();
  }
  if (threadIdx.x == 0) out[b * CO_ + co] = s1[0] / s2[0];
}

// ---------------- host launcher ----------------
extern "C" void kernel_launch(void* const* d_in, const int* in_sizes, int n_in,
                              void* d_out, int out_size, void* d_ws, size_t ws_size,
                              hipStream_t stream) {
  (void)in_sizes; (void)n_in; (void)out_size; (void)ws_size;
  const float* inp  = (const float*)d_in[0];
  const float* L    = (const float*)d_in[1];
  const float* Di   = (const float*)d_in[2];
  const float* DiA  = (const float*)d_in[3];
  const float* mask = (const float*)d_in[4];
  const float* c1W0 = (const float*)d_in[5];
  const float* c1W1 = (const float*)d_in[6];
  const float* c1b  = (const float*)d_in[7];
  const float* bn0g = (const float*)d_in[8];
  const float* bn0b = (const float*)d_in[9];
  const float* fc0W = (const float*)d_in[10];
  const float* fc0b = (const float*)d_in[11];
  const float* bn1g = (const float*)d_in[12];
  const float* bn1b = (const float*)d_in[13];
  const float* fc1W = (const float*)d_in[14];
  const float* fc1b = (const float*)d_in[15];
  const float* g1 = (const float*)d_in[16]; const float* b1 = (const float*)d_in[17];
  const float* g2 = (const float*)d_in[18]; const float* b2 = (const float*)d_in[19];
  const float* g3 = (const float*)d_in[20]; const float* b3 = (const float*)d_in[21];
  const float* c2W0 = (const float*)d_in[22];
  const float* c2W1 = (const float*)d_in[23];
  const float* c2b  = (const float*)d_in[24];

  // workspace carve-up (~292 MB; dominated by bf16 copies of Di/DiA)
  char* w = (char*)d_ws;
  auto alloc = [&](size_t bytes) -> char* {
    char* p = w; w += (bytes + 255) & ~(size_t)255; return p;
  };
  unsigned short* DiB  = (unsigned short*)alloc(2ULL * 8192 * 4096 * 2);
  unsigned short* DiAB = (unsigned short*)alloc(2ULL * 4096 * 8192 * 2);
  unsigned short* LB   = (unsigned short*)alloc(2ULL * 1024 * 1024 * 2);
  float* vb0 = (float*)alloc((size_t)B_ * N_ * C_ * 4);
  float* vb1 = (float*)alloc((size_t)B_ * N_ * C_ * 4);
  float* vb2 = (float*)alloc((size_t)B_ * N_ * C_ * 4);
  float* fb0 = (float*)alloc((size_t)B_ * F__ * C_ * 4);
  float* fb1 = (float*)alloc((size_t)B_ * F__ * C_ * 4);
  float* fb2 = (float*)alloc((size_t)B_ * F__ * C_ * 4);
  float* Dv  = (float*)alloc((size_t)B_ * F__ * C_ * 4);
  float* Df  = (float*)alloc((size_t)B_ * N_ * C_ * 4);
  float* Cpart = (float*)alloc((size_t)B_ * KSPLIT * F__ * C_ * 4);  // split-K partials (max use)
  unsigned short* vqT = (unsigned short*)alloc(2ULL * 32 * 4096 * 2);
  unsigned short* fqT = (unsigned short*)alloc(2ULL * 32 * 8192 * 2);
  unsigned short* uT  = (unsigned short*)alloc(2ULL * 80 * 1024 * 2);
  float* u   = (float*)alloc((size_t)B_ * N_ * C_ * 4);
  float* Lu  = (float*)alloc((size_t)B_ * N_ * C_ * 4);
  float* Lx  = (float*)alloc((size_t)B_ * N_ * CI_ * 4);
  float* vout = (float*)alloc((size_t)B_ * N_ * CO_ * 4);
  float* mu   = (float*)alloc(256 * 4);
  float* rstd = (float*)alloc(256 * 4);
  unsigned short* WTe = (unsigned short*)alloc(80 * 160 * 2);
  float* be = (float*)alloc(128 * 4);

  // one-time (per call) bf16 conversions of the big streaming operands
  k_cvt_bf16<<<4096, 256, 0, stream>>>(Di,  DiB,  2LL * 8192 * 4096);
  k_cvt_bf16<<<4096, 256, 0, stream>>>(DiA, DiAB, 2LL * 4096 * 8192);
  k_cvt_bf16<<<512,  256, 0, stream>>>(L,   LB,   2LL * 1024 * 1024);

  // zero the initial carries (must be redone every call: ws persists across replays)
  k_zero_f32<<<160, 256, 0, stream>>>(vb0, (long)B_ * N_ * C_);
  k_zero_f32<<<320, 256, 0, stream>>>(fb0, (long)B_ * F__ * C_);
  k_zero_f32<<<320, 256, 0, stream>>>(fb1, (long)B_ * F__ * C_);

  // conv1: v = x@W0 + (L@x)@W1 + b
  k_Lx<<<(B_ * N_ * CI_ + 255) / 256, 256, 0, stream>>>(L, inp, Lx);
  k_conv1<<<(B_ * N_ * C_ + 255) / 256, 256, 0, stream>>>(inp, Lx, c1W0, c1W1, c1b, vb1);

  float *vP = vb0, *vC = vb1, *vX = vb2;   // tv0 / v / next
  float *fP = fb0, *fC = fb1, *fX = fb2;   // tmp_f / f / next

  for (int tb = 0; tb < NB_; ++tb) {
    // Dv = Di @ reshape(v)   (skinny bf16 WMMA GEMM, batched, split-K x4)
    k_transpose_bf16<<<dim3(16, 1, 2), 256, 0, stream>>>(vC, vqT, 4096, 20, 32,
                                                         (long)N_ * C_, 32L * 4096);
    k_wmma_gemm<2><<<dim3(64, KSPLIT, 2), 256, 0, stream>>>(DiB, vqT, Cpart,
        4096, 4096 / KSPLIT, 20, 8192L * 4096, 32L * 4096, 8192L * 20);
    k_reduce_k<<<1280, 256, 0, stream>>>(Cpart, Dv, 8192L * 20, KSPLIT);

    // f_new = elu(bn([f|Dv]) @ W0 + c0) + tmp_f  via BN folding
    k_stats<<<160, 256, 0, stream>>>(fC, Dv, B_ * F__, 80, mu, rstd, 0);
    k_fold<<<1, 128, 0, stream>>>(fc0W + (size_t)tb * 160 * 80, bn0g + tb * 160, bn0b + tb * 160,
                                  mu, rstd, fc0b + tb * 80, WTe, be);
    k_wmma_fc<<<B_ * F__ / 128, 256, 0, stream>>>(fC, Dv, WTe, be, fP, fX);

    // Df = DiA @ reshape(f_new)
    k_transpose_bf16<<<dim3(32, 1, 2), 256, 0, stream>>>(fX, fqT, 8192, 20, 32,
                                                         (long)F__ * C_, 32L * 8192);
    k_wmma_gemm<2><<<dim3(32, KSPLIT, 2), 256, 0, stream>>>(DiAB, fqT, Cpart,
        8192, 8192 / KSPLIT, 20, 4096L * 8192, 32L * 8192, 4096L * 20);
    k_reduce_k<<<640, 256, 0, stream>>>(Cpart, Df, 4096L * 20, KSPLIT);

    // v_new = elu(bn([v|Df]) @ W1 + c1) + tv0
    k_stats<<<160, 256, 0, stream>>>(vC, Df, B_ * N_, 80, mu, rstd, 0);
    k_fold<<<1, 128, 0, stream>>>(fc1W + (size_t)tb * 160 * 80, bn1g + tb * 160, bn1b + tb * 160,
                                  mu, rstd, fc1b + tb * 80, WTe, be);
    k_wmma_fc<<<B_ * N_ / 128, 256, 0, stream>>>(vC, Df, WTe, be, vP, vX);

    float* tmp;
    tmp = fP; fP = fC; fC = fX; fX = tmp;
    tmp = vP; vP = vC; vC = vX; vX = tmp;
  }
  // now vC=v2, vP=v1, vX=v0

  k_stats<<<80, 256, 0, stream>>>(vC, vC, B_ * N_, 80, mu, rstd, 0);
  k_stats<<<80, 256, 0, stream>>>(vP, vP, B_ * N_, 80, mu, rstd, 80);
  k_stats<<<80, 256, 0, stream>>>(vX, vX, B_ * N_, 80, mu, rstd, 160);
  k_merge<<<(B_ * N_ * C_ + 255) / 256, 256, 0, stream>>>(vC, vP, vX, mu, rstd,
                                                          g1, b1, g2, b2, g3, b3, u);
  // conv2: Lu = L @ u via WMMA (NT=5, split-K), then small fused projection + masked pool
  k_transpose_bf16<<<dim3(4, 1, 2), 256, 0, stream>>>(u, uT, 1024, 80, 80,
                                                      (long)N_ * C_, 80L * 1024);
  k_wmma_gemm<5><<<dim3(8, KSPLIT, 2), 256, 0, stream>>>(LB, uT, Cpart,
      1024, 1024 / KSPLIT, 80, 1024L * 1024, 80L * 1024, 1024L * 80);
  k_reduce_k<<<640, 256, 0, stream>>>(Cpart, Lu, 1024L * 80, KSPLIT);

  k_conv2<<<(B_ * N_ * CO_ + 255) / 256, 256, 0, stream>>>(u, Lu, c2W0, c2W1, c2b, vout);
  k_pool<<<dim3(CO_, B_), 256, 0, stream>>>(vout, mask, (float*)d_out);
}
